// S4D_240518169035
// MI455X (gfx1250) — matrix operand order repaired and checked by reading, not compiled, not run on hardware
//
#include <hip/hip_runtime.h>
#include <stdint.h>

#define HH 128
#define NN 64
#define LL 2048
#define BB 2
#define TT 16           // chunk length
#define NCHUNK (LL/TT)  // 128

typedef __attribute__((ext_vector_type(16))) __bf16 v16bf;
typedef __attribute__((ext_vector_type(8)))  float  v8f;

__device__ inline float2 cmul(float2 a, float2 b) {
    return make_float2(a.x*b.x - a.y*b.y, a.x*b.y + a.y*b.x);
}
__device__ inline float2 cadd(float2 a, float2 b) {
    return make_float2(a.x + b.x, a.y + b.y);
}
__device__ inline uint16_t f32_to_bf16(float x) {
    uint32_t u = __float_as_uint(x);
    uint32_t r = u + 0x7FFFu + ((u >> 16) & 1u);   // round-to-nearest-even
    return (uint16_t)(r >> 16);
}
__device__ inline float bf16_to_f32(uint16_t h) {
    return __uint_as_float(((uint32_t)h) << 16);
}
__device__ inline __bf16 to_bf16(float x) {
    union { uint16_t u; __bf16 h; } cv;
    cv.u = f32_to_bf16(x);
    return cv.h;
}
__device__ inline v8f wmma_bf16(v16bf a, v16bf b, v8f c) {
    return __builtin_amdgcn_wmma_f32_16x16x32_bf16(false, a, false, b,
                                                   (short)0, c, false, false);
}

// 16-bit A-operand k order used in this file (16x32, contiguous; B uses the same):
//   lane<16 : M=lane,    halfword idx -> K = idx
//   lane>=16: M=lane-16, halfword idx -> K = 16+idx
__device__ inline int a_k_of(int lane, int idx) {
    return idx + ((lane >= 16) ? 16 : 0);
}

// ---------------------------------------------------------------------------
// Kernel 1: per-h operand packing.
//  - A_tri : 16x16 lower-tri Toeplitz of K[t]=Re(C·dB·e^{dA t}) (K-dim padded)
//  - A_E   : 8 frags (4 n-tiles x re/im) of E[n][j] = dB[n]·e^{dA(15-j)} (K pad)
//  - A_W   : 4 frags (2 n-halves x {Re, -Im}) of W[i][n] = C[n]·e^{dA(i+1)}, K=32 full
//  - P     : e^{16·dA} per (h,n) for the chunk recurrence
// grid = H blocks, block = N threads.
// ---------------------------------------------------------------------------
__global__ void s4d_params_kernel(const float* __restrict__ log_A_real,
                                  const float* __restrict__ A_imag,
                                  const float* __restrict__ Bp,
                                  const float* __restrict__ log_dt,
                                  const float* __restrict__ Cp,
                                  uint16_t* __restrict__ ApackTri,
                                  uint16_t* __restrict__ ApackE,
                                  uint16_t* __restrict__ ApackW,
                                  float*    __restrict__ Pst) {
    __shared__ float kpart[NN][TT];
    __shared__ float Ksh[TT];
    __shared__ float EreS[NN][TT], EimS[NN][TT];
    __shared__ float WreS[TT][NN], WimS[TT][NN];

    const int h = blockIdx.x;
    const int n = threadIdx.x;
    const int hn = h * NN + n;

    const float step = __expf(log_dt[0]);
    const float lamr = -__expf(log_A_real[hn]);
    const float lami = A_imag[hn];
    const float dAr = lamr * step, dAi = lami * step;
    const float edr = __expf(dAr);
    const float2 eA = make_float2(edr * __cosf(dAi), edr * __sinf(dAi));

    // dB = Bc * (exp(dA) - 1) / Lam
    const float2 em1 = make_float2(eA.x - 1.f, eA.y);
    const float inv = 1.f / (lamr * lamr + lami * lami);
    const float2 t  = make_float2((em1.x * lamr + em1.y * lami) * inv,
                                  (em1.y * lamr - em1.x * lami) * inv);
    const float2 Bc = make_float2(Bp[hn * 2], Bp[hn * 2 + 1]);
    const float2 dB = cmul(Bc, t);
    const float2 Cc = make_float2(Cp[n * 2], Cp[n * 2 + 1]);
    const float2 w  = cmul(Cc, dB);

    float2 P = eA;                        // e^{16 dA} = eA^16
    #pragma unroll
    for (int i = 0; i < 4; ++i) P = cmul(P, P);
    Pst[hn * 2]     = P.x;
    Pst[hn * 2 + 1] = P.y;

    // K[t] partials (intra Toeplitz kernel)
    {
        float2 f = w;
        #pragma unroll
        for (int tt = 0; tt < TT; ++tt) { kpart[n][tt] = f.x; f = cmul(f, eA); }
    }
    // E[n][j] = dB * e^{dA (15-j)}
    {
        float2 f = dB;
        #pragma unroll
        for (int j = TT - 1; j >= 0; --j) { EreS[n][j] = f.x; EimS[n][j] = f.y; f = cmul(f, eA); }
    }
    // W[i][n] = Cc * e^{dA (i+1)}
    {
        float2 f = cmul(Cc, eA);
        #pragma unroll
        for (int i = 0; i < TT; ++i) { WreS[i][n] = f.x; WimS[i][n] = f.y; f = cmul(f, eA); }
    }
    __syncthreads();
    if (n < TT) {
        float acc = 0.f;
        #pragma unroll
        for (int m = 0; m < NN; ++m) acc += kpart[m][n];
        Ksh[n] = acc;
    }
    __syncthreads();

    // Pack A_tri (512 entries, 8/thread)
    #pragma unroll
    for (int j = 0; j < 8; ++j) {
        const int e = n * 8 + j;
        const int lane = e >> 4, idx = e & 15;
        const int m = lane & 15;
        const int k = a_k_of(lane, idx);
        float v = 0.f;
        if (k < TT && m >= k) v = Ksh[m - k];
        ApackTri[h * 512 + e] = f32_to_bf16(v);
    }
    // Pack E fragments: frag f = tile*2 + isIm; rows m = n_local, K = j (pad >=16)
    for (int f = 0; f < 8; ++f) {
        const int tile = f >> 1, isim = f & 1;
        #pragma unroll
        for (int j = 0; j < 8; ++j) {
            const int e = n * 8 + j;
            const int lane = e >> 4, idx = e & 15;
            const int m = lane & 15;
            const int k = a_k_of(lane, idx);
            float v = 0.f;
            if (k < TT) v = isim ? EimS[tile * TT + m][k] : EreS[tile * TT + m][k];
            ApackE[(h * 8 + f) * 512 + e] = f32_to_bf16(v);
        }
    }
    // Pack W fragments: frag f = half*2 + isIm; rows m = i, K = n_local (full 32)
    for (int f = 0; f < 4; ++f) {
        const int half = f >> 1, isim = f & 1;
        #pragma unroll
        for (int j = 0; j < 8; ++j) {
            const int e = n * 8 + j;
            const int lane = e >> 4, idx = e & 15;
            const int m = lane & 15;
            const int k = a_k_of(lane, idx);
            const int nn2 = half * 32 + k;
            const float v = isim ? -WimS[m][nn2] : WreS[m][nn2];
            ApackW[(h * 4 + f) * 512 + e] = f32_to_bf16(v);
        }
    }
}

// ---------------------------------------------------------------------------
// Kernel 2: transpose u[B,L,H] f32 -> ubf[B,H,L] bf16 (coalesced reads).
// ---------------------------------------------------------------------------
__global__ void s4d_transpose_kernel(const float* __restrict__ u,
                                     uint16_t* __restrict__ ubf) {
    const int idx = blockIdx.x * 256 + threadIdx.x;   // B*L*H total
    const int b = idx / (LL * HH);
    const int r = idx % (LL * HH);
    const int l = r / HH;
    const int h = r % HH;
    ubf[(b * HH + h) * LL + l] = f32_to_bf16(u[idx]);
}

// ---------------------------------------------------------------------------
// Kernel 3: fused chunked scan, one workgroup (4 waves) per (b,h).
// Phase 1: intra Toeplitz conv (regs) + Q = E x u  -> LDS     [WMMA]
// Phase 2: 128-step chunk recurrence s' = P*s + Q  (wave 0)   [VALU]
// Phase 3: y_state = Re(W x S), combine, store                [WMMA]
// ---------------------------------------------------------------------------
__global__ void __launch_bounds__(128)
s4d_fused_kernel(const float* __restrict__ u,
                 const float* __restrict__ Dvec,
                 const float* __restrict__ Pst,
                 const uint16_t* __restrict__ ApackTri,
                 const uint16_t* __restrict__ ApackE,
                 const uint16_t* __restrict__ ApackW,
                 const uint16_t* __restrict__ ubf,
                 float* __restrict__ out) {
    __shared__ float Qre[NN * NCHUNK];   // 32 KB
    __shared__ float Qim[NN * NCHUNK];   // 32 KB

    const int bh = blockIdx.x;
    const int h  = bh & (HH - 1);
    const int b  = bh >> 7;
    const int tid = threadIdx.x, wave = tid >> 5, lane = tid & 31;
    const int col = lane & 15, hi = lane >> 4;

    // ---- Phase 1: intra conv + state-input projection -------------------
    v16bf a_tri = *reinterpret_cast<const v16bf*>(ApackTri + h * 512 + lane * 16);
    v16bf aE[8];
    #pragma unroll
    for (int f = 0; f < 8; ++f)
        aE[f] = *reinterpret_cast<const v16bf*>(ApackE + (h * 8 + f) * 512 + lane * 16);

    __builtin_prefetch(ubf + bh * LL + (wave * 2 + 1) * 256, 0, 1);

    v8f yin[2];
    #pragma unroll
    for (int gg = 0; gg < 2; ++gg) {
        const int g = wave * 2 + gg;           // chunk-group: 16 chunks, 256 pos
        v16bf bfrag = {};
        if (lane < 16)  // lane n = chunk col; K = 16 positions; lanes>=16 = K pad
            bfrag = *reinterpret_cast<const v16bf*>(ubf + bh * LL + g * 256 + lane * 16);

        v8f c0 = {};
        c0 = wmma_bf16(a_tri, bfrag, c0);      // y_intra, kept in registers
        yin[gg] = c0;

        #pragma unroll
        for (int tile = 0; tile < 4; ++tile) { // Q[n][c], n-tile of 16
            v8f cr = {}, ci = {};
            cr = wmma_bf16(aE[tile * 2 + 0], bfrag, cr);
            ci = wmma_bf16(aE[tile * 2 + 1], bfrag, ci);
            #pragma unroll
            for (int r = 0; r < 8; ++r) {
                const int nidx = tile * TT + hi * 8 + r;
                Qre[nidx * NCHUNK + g * 16 + col] = cr[r];
                Qim[nidx * NCHUNK + g * 16 + col] = ci[r];
            }
        }
    }
    __syncthreads();

    // ---- Phase 2: chunk-level recurrence (only sequential part) ---------
    if (wave == 0) {
        const int n1 = lane, n2 = lane + 32;
        const float2 P1 = make_float2(Pst[(h * NN + n1) * 2], Pst[(h * NN + n1) * 2 + 1]);
        const float2 P2 = make_float2(Pst[(h * NN + n2) * 2], Pst[(h * NN + n2) * 2 + 1]);
        float2 s1 = make_float2(0.f, 0.f), s2 = make_float2(0.f, 0.f);
        for (int c = 0; c < NCHUNK; ++c) {
            const float2 q1 = make_float2(Qre[n1 * NCHUNK + c], Qim[n1 * NCHUNK + c]);
            const float2 q2 = make_float2(Qre[n2 * NCHUNK + c], Qim[n2 * NCHUNK + c]);
            Qre[n1 * NCHUNK + c] = s1.x; Qim[n1 * NCHUNK + c] = s1.y;  // S_c (pre-chunk)
            Qre[n2 * NCHUNK + c] = s2.x; Qim[n2 * NCHUNK + c] = s2.y;
            s1 = cadd(cmul(P1, s1), q1);
            s2 = cadd(cmul(P2, s2), q2);
        }
    }
    __syncthreads();

    // ---- Phase 3: y_state = Re(W x S) + combine -------------------------
    v16bf wfrag[4];
    #pragma unroll
    for (int f = 0; f < 4; ++f)
        wfrag[f] = *reinterpret_cast<const v16bf*>(ApackW + (h * 4 + f) * 512 + lane * 16);
    const float Dh = Dvec[h];

    #pragma unroll
    for (int gg = 0; gg < 2; ++gg) {
        const int g = wave * 2 + gg;
        v8f acc = {};
        #pragma unroll
        for (int half = 0; half < 2; ++half) {  // contraction over n: 2 x K=32
            v16bf bre = {}, bim = {};
            #pragma unroll
            for (int i = 0; i < 16; ++i) {
                const int nidx = half * 32 + hi * 16 + i;   // B layout: K = hi*16+i
                bre[i] = to_bf16(Qre[nidx * NCHUNK + g * 16 + col]);
                bim[i] = to_bf16(Qim[nidx * NCHUNK + g * 16 + col]);
            }
            acc = wmma_bf16(wfrag[half * 2 + 0], bre, acc);  //  Re(W)*Re(S)
            acc = wmma_bf16(wfrag[half * 2 + 1], bim, acc);  // -Im(W)*Im(S)
        }
        const int lbase = (g * 16 + col) * TT + hi * 8;
        #pragma unroll
        for (int r = 0; r < 8; ++r) {
            const int l = lbase + r;
            const int oi = (b * LL + l) * HH + h;
            out[oi] = acc[r] + yin[gg][r] + Dh * u[oi];
        }
    }
}

extern "C" void kernel_launch(void* const* d_in, const int* in_sizes, int n_in,
                              void* d_out, int out_size, void* d_ws, size_t ws_size,
                              hipStream_t stream) {
    const float* u          = (const float*)d_in[0]; // [B,L,H]
    const float* log_A_real = (const float*)d_in[1]; // [H,N]
    const float* A_imag     = (const float*)d_in[2]; // [H,N]
    const float* Bp         = (const float*)d_in[3]; // [H,N,2]
    const float* log_dt     = (const float*)d_in[4]; // [1]
    const float* Cp         = (const float*)d_in[5]; // [1,N,2]
    const float* Dv         = (const float*)d_in[6]; // [H]
    float* out = (float*)d_out;

    // Workspace layout (all offsets 4KB-aligned):
    //   ApackTri : H*512 bf16        = 131072 B   @ 0
    //   ApackE   : H*8*512 bf16      = 1048576 B  @ 131072
    //   ApackW   : H*4*512 bf16      = 524288 B   @ 1179648
    //   Pst      : H*N*2 f32         = 65536 B    @ 1703936
    //   ubf      : B*H*L bf16        = 1048576 B  @ 1769472
    char* ws = (char*)d_ws;
    uint16_t* ApackTri = (uint16_t*)(ws);
    uint16_t* ApackE   = (uint16_t*)(ws + 131072);
    uint16_t* ApackW   = (uint16_t*)(ws + 1179648);
    float*    Pst      = (float*)   (ws + 1703936);
    uint16_t* ubf      = (uint16_t*)(ws + 1769472);

    s4d_params_kernel<<<HH, NN, 0, stream>>>(log_A_real, A_imag, Bp, log_dt, Cp,
                                             ApackTri, ApackE, ApackW, Pst);
    s4d_transpose_kernel<<<(BB * LL * HH) / 256, 256, 0, stream>>>(u, ubf);
    s4d_fused_kernel<<<BB * HH, 128, 0, stream>>>(u, Dv, Pst, ApackTri, ApackE,
                                                  ApackW, ubf, out);
}
